// MultiHeadAttention_53085795778890
// MI455X (gfx1250) — matrix operand order repaired and checked by reading, not compiled, not run on hardware
//
#include <hip/hip_runtime.h>

#define BB 2
#define NN 2048
#define CC 1024
#define HH 16
#define DD 64
#define C3 3072

typedef __attribute__((ext_vector_type(16))) __bf16 v16bf;
typedef __attribute__((ext_vector_type(8)))  float  v8f;

__device__ __forceinline__ unsigned short f32_to_bf16(float f) {
  unsigned int u = __float_as_uint(f);
  u += 0x7fffu + ((u >> 16) & 1u);   // round-to-nearest-even
  return (unsigned short)(u >> 16);
}

__device__ __forceinline__ v8f v8f_zero() {
  v8f z = {0.f, 0.f, 0.f, 0.f, 0.f, 0.f, 0.f, 0.f};
  return z;
}

// 16-bit A-matrix 16x32 fragment K pattern (ISA 7.12.2):
// lanes 0-15: v0..v3 -> K 0..7, v4..v7 -> K 16..23 ; lanes 16-31: +8
__device__ __forceinline__ int kA0(int j, int half) {
  return ((j >> 2) << 4) + half * 8 + ((j & 3) << 1);
}
// 16-bit B-matrix 32x16 fragment K pattern (same as A):
// lanes 0-15: v0..v3 -> K 0..7, v4..v7 -> K 16..23 ; lanes 16-31: +8
__device__ __forceinline__ int kB0(int j, int half) {
  return ((j >> 2) << 4) + half * 8 + ((j & 3) << 1);
}

__device__ __forceinline__ v16bf load_fragA(const unsigned short* __restrict__ p,
                                            int row, int stride, int kbase, int half) {
  union { v16bf v; unsigned int u[8]; } f;
  const unsigned short* b = p + row * stride + kbase;
#pragma unroll
  for (int j = 0; j < 8; ++j)
    f.u[j] = *reinterpret_cast<const unsigned int*>(b + kA0(j, half));
  return f.v;
}

__device__ __forceinline__ v16bf load_fragB(const unsigned short* __restrict__ p,
                                            int col, int stride, int kbase, int half) {
  union { v16bf v; unsigned int u[8]; } f;
  const unsigned short* b = p + col * stride + kbase;
#pragma unroll
  for (int j = 0; j < 8; ++j)
    f.u[j] = *reinterpret_cast<const unsigned int*>(b + kB0(j, half));
  return f.v;
}

__device__ __forceinline__ v8f wmma_bf16(v16bf a, v16bf b, v8f c) {
  return __builtin_amdgcn_wmma_f32_16x16x32_bf16(false, a, false, b, (short)0, c,
                                                 false, false);
}

// ---------------- elementwise convert / transpose ----------------

__global__ void cvt_f32_bf16(const float* __restrict__ in,
                             unsigned short* __restrict__ out, int n) {
  int i = blockIdx.x * blockDim.x + threadIdx.x;
  if (i < n) out[i] = f32_to_bf16(in[i]);
}

// out[n*K + k] = bf16(in[k*N + n])   (in: [K][N] f32, out: [N][K] bf16)
__global__ void transpose_bf16(const float* __restrict__ in,
                               unsigned short* __restrict__ out, int K, int N) {
  int i = blockIdx.x * blockDim.x + threadIdx.x;
  if (i >= K * N) return;
  int n = i / K, k = i - n * K;
  out[i] = f32_to_bf16(in[k * N + n]);
}

// ---------------- shared GEMM core: 32x64 tile, double-buffered ----------------

struct FragSet {
  v16bf a[2];   // two 16-row A tiles
  v16bf b[4];   // four 16-col B tiles
};

__device__ __forceinline__ void load_set(FragSet& f,
                                         const unsigned short* __restrict__ A,
                                         const unsigned short* __restrict__ BT,
                                         int m0, int ncol, int lo, int half,
                                         int kb, int K) {
  f.a[0] = load_fragA(A, m0 + lo, K, kb, half);
  f.a[1] = load_fragA(A, m0 + 16 + lo, K, kb, half);
#pragma unroll
  for (int t = 0; t < 4; ++t)
    f.b[t] = load_fragB(BT, ncol + t * 16 + lo, K, kb, half);
}

__device__ __forceinline__ void gemm_core(const unsigned short* __restrict__ A,
                                          const unsigned short* __restrict__ BT,
                                          int m0, int ncol, int lo, int half, int K,
                                          v8f (&acc)[2][4]) {
#pragma unroll
  for (int r = 0; r < 2; ++r)
#pragma unroll
    for (int t = 0; t < 4; ++t) acc[r][t] = v8f_zero();

  FragSet f0, f1;
  load_set(f0, A, BT, m0, ncol, lo, half, 0, K);

  auto compute = [&](FragSet& f) {
#pragma unroll
    for (int r = 0; r < 2; ++r)
#pragma unroll
      for (int t = 0; t < 4; ++t)
        acc[r][t] = wmma_bf16(f.a[r], f.b[t], acc[r][t]);
  };

  // K/32 blocks (even count). Ping-pong: issue loads for the next block(s)
  // before computing the current one, so waits can be partial.
#pragma unroll 1
  for (int kb = 32; kb + 64 <= K; kb += 64) {
    load_set(f1, A, BT, m0, ncol, lo, half, kb, K);
    compute(f0);
    load_set(f0, A, BT, m0, ncol, lo, half, kb + 32, K);
    compute(f1);
  }
  compute(f0);   // last computed (K/32 even -> f0 holds block K-64; block K-32 never loaded)
}

// ---------------- QKV projection GEMM ----------------
// xb: [B*N][C] bf16, wT: [3C][C] bf16 (N-major), bias: [3C] f32
// Q,Km: [B*H][N][D] bf16 ; Vt: [B*H][D][N] bf16
__global__ void __launch_bounds__(256) gemm_qkv(
    const unsigned short* __restrict__ xb,
    const unsigned short* __restrict__ wT,
    const float* __restrict__ bias,
    unsigned short* __restrict__ Q,
    unsigned short* __restrict__ Km,
    unsigned short* __restrict__ Vt) {
  int wave = blockIdx.x * 8 + (threadIdx.x >> 5);
  int lane = threadIdx.x & 31, half = lane >> 4, lo = lane & 15;
  const int ntiles = C3 / 64;                       // 48
  int mt = wave / ntiles, ncol = (wave - mt * ntiles) * 64;
  int m0 = mt * 32;

  v8f acc[2][4];
  gemm_core(xb, wT, m0, ncol, lo, half, CC, acc);

#pragma unroll
  for (int t = 0; t < 4; ++t) {
    int n = ncol + t * 16 + lo;
    float bv = bias[n];
    int sec = n >> 10, cc = n & 1023, h = cc >> 6, dd = cc & 63;
#pragma unroll
    for (int r = 0; r < 2; ++r)
#pragma unroll
      for (int i = 0; i < 8; ++i) {
        int row = m0 + r * 16 + half * 8 + i;
        int bidx = row >> 11, nrow = row & 2047;
        unsigned short v = f32_to_bf16(acc[r][t][i] + bv);
        int bh = bidx * HH + h;
        if (sec == 0)      Q [(bh * NN + nrow) * DD + dd] = v;
        else if (sec == 1) Km[(bh * NN + nrow) * DD + dd] = v;
        else               Vt[(bh * DD + dd) * NN + nrow] = v;
      }
  }
}

// ---------------- flash attention ----------------
// Q,Km: [B*H][N][D] bf16 ; Vt: [B*H][D][N] bf16 ; out: [B*N][C] bf16
__global__ void __launch_bounds__(256) attn_kernel(
    const unsigned short* __restrict__ Q,
    const unsigned short* __restrict__ Km,
    const unsigned short* __restrict__ Vt,
    unsigned short* __restrict__ out) {
  __shared__ unsigned short ps[8][16 * 32];          // per-wave P staging
  int wv = threadIdx.x >> 5;
  int wave = blockIdx.x * 8 + wv;
  int lane = threadIdx.x & 31, half = lane >> 4, lo = lane & 15;
  int bh = wave >> 7;                                // 128 q-tiles per (b,h)
  int qt = wave & 127;
  int b = bh >> 4, h = bh & 15;
  const unsigned short* Qb = Q  + bh * (NN * DD);
  const unsigned short* Kb = Km + bh * (NN * DD);
  const unsigned short* Vb = Vt + bh * (DD * NN);
  int q0 = qt * 16;

  v16bf qa0 = load_fragA(Qb, q0 + lo, DD, 0, half);
  v16bf qa1 = load_fragA(Qb, q0 + lo, DD, 32, half);

  float mrun[8], lrun[8];
  v8f acc[4];
#pragma unroll
  for (int i = 0; i < 8; ++i) { mrun[i] = -3.0e38f; lrun[i] = 0.f; }
#pragma unroll
  for (int t = 0; t < 4; ++t) acc[t] = v8f_zero();

  const float scale = 0.125f;                        // 1/sqrt(64)

  auto load_kf = [&](v16bf (&kf)[4], int kb) {
    kf[0] = load_fragB(Kb, kb + lo,      DD, 0,  half);
    kf[1] = load_fragB(Kb, kb + lo,      DD, 32, half);
    kf[2] = load_fragB(Kb, kb + 16 + lo, DD, 0,  half);
    kf[3] = load_fragB(Kb, kb + 16 + lo, DD, 32, half);
  };

  // One 32-key block: scores from prefetched K frags, prefetch V (this block)
  // and K (next block) before the softmax VALU work so L2 latency hides.
  auto block = [&](int kb, v16bf (&kfc)[4], v16bf (&kfn)[4]) {
    v8f s0 = v8f_zero(), s1 = v8f_zero();
    s0 = wmma_bf16(qa0, kfc[0], s0);
    s0 = wmma_bf16(qa1, kfc[1], s0);
    s1 = wmma_bf16(qa0, kfc[2], s1);
    s1 = wmma_bf16(qa1, kfc[3], s1);

    v16bf vf[4];
#pragma unroll
    for (int t = 0; t < 4; ++t)
      vf[t] = load_fragB(Vb, t * 16 + lo, NN, kb, half);
    load_kf(kfn, (kb + 32) & (NN - 1));              // branchless next-K prefetch

    float alpha[8];
#pragma unroll
    for (int i = 0; i < 8; ++i) {
      float a0 = s0[i] * scale, a1 = s1[i] * scale;
      float rm = fmaxf(a0, a1);
      rm = fmaxf(rm, __shfl_xor(rm, 1, 32));
      rm = fmaxf(rm, __shfl_xor(rm, 2, 32));
      rm = fmaxf(rm, __shfl_xor(rm, 4, 32));
      rm = fmaxf(rm, __shfl_xor(rm, 8, 32));
      float mn = fmaxf(mrun[i], rm);
      float al = __expf(mrun[i] - mn);
      float p0 = __expf(a0 - mn);
      float p1 = __expf(a1 - mn);
      float rs = p0 + p1;
      rs += __shfl_xor(rs, 1, 32);
      rs += __shfl_xor(rs, 2, 32);
      rs += __shfl_xor(rs, 4, 32);
      rs += __shfl_xor(rs, 8, 32);
      lrun[i] = lrun[i] * al + rs;
      mrun[i] = mn;
      alpha[i] = al;
      int m = half * 8 + i;
      ps[wv][m * 32 + lo]      = f32_to_bf16(p0);    // D-layout -> row-major LDS
      ps[wv][m * 32 + 16 + lo] = f32_to_bf16(p1);
    }
#pragma unroll
    for (int t = 0; t < 4; ++t)
#pragma unroll
      for (int i = 0; i < 8; ++i) acc[t][i] *= alpha[i];

    asm volatile("s_wait_dscnt 0" ::: "memory");     // DS in-order per wave

    union { v16bf v; unsigned int u[8]; } pf;        // re-gather as A fragment
#pragma unroll
    for (int j = 0; j < 8; ++j)
      pf.u[j] = *reinterpret_cast<const unsigned int*>(&ps[wv][lo * 32 + kA0(j, half)]);

#pragma unroll
    for (int t = 0; t < 4; ++t)
      acc[t] = wmma_bf16(pf.v, vf[t], acc[t]);
  };

  v16bf kf0[4], kf1[4];
  load_kf(kf0, 0);
#pragma unroll 1
  for (int kb = 0; kb < NN; kb += 64) {              // NN/32 even -> pairs
    block(kb, kf0, kf1);
    block(kb + 32, kf1, kf0);
  }

#pragma unroll
  for (int t = 0; t < 4; ++t)
#pragma unroll
    for (int i = 0; i < 8; ++i) {
      int m = half * 8 + i;
      float v = acc[t][i] / lrun[i];
      out[(b * NN + q0 + m) * CC + h * DD + t * 16 + lo] = f32_to_bf16(v);
    }
}

// ---------------- output projection GEMM ----------------
// ab: [B*N][C] bf16, wT: [C][C] bf16 (N-major), bias: [C] f32, out: [B*N][C] f32
__global__ void __launch_bounds__(256) gemm_out(
    const unsigned short* __restrict__ ab,
    const unsigned short* __restrict__ wT,
    const float* __restrict__ bias,
    float* __restrict__ out) {
  int wave = blockIdx.x * 8 + (threadIdx.x >> 5);
  int lane = threadIdx.x & 31, half = lane >> 4, lo = lane & 15;
  const int ntiles = CC / 64;                        // 16
  int mt = wave / ntiles, ncol = (wave - mt * ntiles) * 64;
  int m0 = mt * 32;

  v8f acc[2][4];
  gemm_core(ab, wT, m0, ncol, lo, half, CC, acc);

#pragma unroll
  for (int t = 0; t < 4; ++t) {
    int n = ncol + t * 16 + lo;
    float bv = bias[n];
#pragma unroll
    for (int r = 0; r < 2; ++r)
#pragma unroll
      for (int i = 0; i < 8; ++i)
        out[(m0 + r * 16 + half * 8 + i) * CC + n] = acc[r][t][i] + bv;
  }
}

extern "C" void kernel_launch(void* const* d_in, const int* in_sizes, int n_in,
                              void* d_out, int out_size, void* d_ws, size_t ws_size,
                              hipStream_t stream) {
  const float* x    = (const float*)d_in[0];
  const float* Wqkv = (const float*)d_in[1];
  const float* bqkv = (const float*)d_in[2];
  const float* Wout = (const float*)d_in[3];
  const float* bout = (const float*)d_in[4];

  char* ws = (char*)d_ws;
  size_t off = 0;
  auto alloc = [&](size_t bytes) -> void* {
    void* p = ws + off;
    off += (bytes + 255) & ~size_t(255);
    return p;
  };
  unsigned short* xb   = (unsigned short*)alloc(size_t(BB) * NN * CC * 2);  // 8 MB
  unsigned short* wqT  = (unsigned short*)alloc(size_t(C3) * CC * 2);       // 6 MB
  unsigned short* woT  = (unsigned short*)alloc(size_t(CC) * CC * 2);       // 2 MB
  unsigned short* Qd   = (unsigned short*)alloc(size_t(BB) * HH * NN * DD * 2);
  unsigned short* Kd   = (unsigned short*)alloc(size_t(BB) * HH * NN * DD * 2);
  unsigned short* Vtd  = (unsigned short*)alloc(size_t(BB) * HH * DD * NN * 2);
  unsigned short* attn = (unsigned short*)alloc(size_t(BB) * NN * CC * 2);  // 8 MB

  int nx = BB * NN * CC;
  cvt_f32_bf16<<<(nx + 255) / 256, 256, 0, stream>>>(x, xb, nx);
  transpose_bf16<<<(CC * C3 + 255) / 256, 256, 0, stream>>>(Wqkv, wqT, CC, C3);
  transpose_bf16<<<(CC * CC + 255) / 256, 256, 0, stream>>>(Wout, woT, CC, CC);

  gemm_qkv<<<(BB * NN / 32) * (C3 / 64) / 8, 256, 0, stream>>>(xb, wqT, bqkv, Qd, Kd, Vtd);
  attn_kernel<<<(BB * HH * (NN / 16)) / 8, 256, 0, stream>>>(Qd, Kd, Vtd, attn);
  gemm_out<<<(BB * NN / 32) * (CC / 64) / 8, 256, 0, stream>>>(attn, woT, bout, (float*)d_out);
}
